// A100SimilarityCorrector_58007828300466
// MI455X (gfx1250) — matrix operand order repaired and checked by reading, not compiled, not run on hardware
//
#include <hip/hip_runtime.h>
#include <hip/hip_bf16.h>
#include <math.h>

typedef __attribute__((ext_vector_type(16))) _Float16 v16h;
typedef __attribute__((ext_vector_type(8)))  _Float16 v8h;
typedef __attribute__((ext_vector_type(8)))  float    v8f;

#define EPSV 1e-5f
#define NN   512
#define BB   8
#define NELEM (BB*NN*NN)          // 2,097,152
#define NTILE (NELEM/16)          // 131,072 16-element tiles
#define WGS   1024                // workgroups for kernel 1

// ---------------------------------------------------------------------------
// Kernel 1: per-element 4-layer MLP. 16 elements per wave-iteration mapped to
// the M dim of v_wmma_f32_16x16x32_f16. BatchNorm folded into W2/W3.
// Writes the pre-symmetrization result to out[e].
// ---------------------------------------------------------------------------
__global__ __launch_bounds__(256)
void mlp_wmma_kernel(const float* __restrict__ sim,
                     const int*   __restrict__ nmask,
                     const float* __restrict__ W1, const float* __restrict__ b1,
                     const float* __restrict__ g1, const float* __restrict__ be1,
                     const float* __restrict__ m1, const float* __restrict__ v1,
                     const float* __restrict__ W2, const float* __restrict__ b2,
                     const float* __restrict__ g2, const float* __restrict__ be2,
                     const float* __restrict__ m2, const float* __restrict__ v2,
                     const float* __restrict__ W3, const float* __restrict__ b3,
                     const float* __restrict__ W4, const float* __restrict__ b4,
                     const float* __restrict__ rw_p,
                     float* __restrict__ out)
{
    __shared__ __align__(16) _Float16 lds[8][16 * 64];   // 2KB per wave, 16KB/WG

    const int tid  = threadIdx.x;
    const int wave = tid >> 5;
    const int lane = tid & 31;
    const int hi   = lane >> 4;    // which 16-lane half
    const int col  = lane & 15;    // N-index in B/C/D layout, M-row in A layout
    _Float16* my = &lds[wave][0];

    // ---- one-time per-lane setup -------------------------------------------
    // Layer-1 coefficients in A-register layout order (halves i=0..7 -> K=klo+i,
    // halves 8..15 -> K=klo+16+i, klo = 8*hi; tile T adds 32).
    _Float16 w1h[2][16], b1h[2][16];
#pragma unroll
    for (int T = 0; T < 2; ++T)
#pragma unroll
        for (int i = 0; i < 16; ++i) {
            int K = 32 * T + 8 * hi + (i & 7) + ((i >> 3) << 4);
            w1h[T][i] = (_Float16)W1[K];
            b1h[T][i] = (_Float16)b1[K];
        }

    // BN1 folded into W2:  W2'[k][n] = s1[k] * W2[k][n]
    // B layout (32x16 f16): half i -> K = 8*hi + (i&7) + 16*(i>>3), column = col.
    v16h Bt2[2][4];
#pragma unroll
    for (int kb = 0; kb < 2; ++kb)
#pragma unroll
        for (int i = 0; i < 16; ++i) {
            int k = 32 * kb + 8 * hi + (i & 7) + ((i >> 3) << 4);
            float s = g1[k] * rsqrtf(v1[k] + EPSV);
#pragma unroll
            for (int nb = 0; nb < 4; ++nb)
                Bt2[kb][nb][i] = (_Float16)(s * W2[k * 64 + 16 * nb + col]);
        }

    // BN2 folded into W3
    v16h Bt3[2][2];
#pragma unroll
    for (int kb = 0; kb < 2; ++kb)
#pragma unroll
        for (int i = 0; i < 16; ++i) {
            int k = 32 * kb + 8 * hi + (i & 7) + ((i >> 3) << 4);
            float s = g2[k] * rsqrtf(v2[k] + EPSV);
#pragma unroll
            for (int nb = 0; nb < 2; ++nb)
                Bt3[kb][nb][i] = (_Float16)(s * W3[k * 32 + 16 * nb + col]);
        }

    // Folded biases: b2' = b2 + t1@W2, b3' = b3 + t2@W3 (per lane: its columns)
    float b2p[4] = { b2[col], b2[16 + col], b2[32 + col], b2[48 + col] };
    float b3p[2] = { b3[col], b3[16 + col] };
    for (int k = 0; k < 64; ++k) {
        float s1k = g1[k] * rsqrtf(v1[k] + EPSV);
        float t1k = be1[k] - s1k * m1[k];
        float s2k = g2[k] * rsqrtf(v2[k] + EPSV);
        float t2k = be2[k] - s2k * m2[k];
#pragma unroll
        for (int nb = 0; nb < 4; ++nb) b2p[nb] += t1k * W2[k * 64 + 16 * nb + col];
#pragma unroll
        for (int nb = 0; nb < 2; ++nb) b3p[nb] += t2k * W3[k * 32 + 16 * nb + col];
    }

    const float w4a = W4[col];
    const float w4b = W4[16 + col];
    const float b4s = b4[0];
    const float rw  = rw_p[0];

    // ---- main loop: one 16-element tile per wave iteration -----------------
    const int nwaves = gridDim.x * (blockDim.x >> 5);
    for (int t = blockIdx.x * (blockDim.x >> 5) + wave; t < NTILE; t += nwaves) {
        const float x = sim[t * 16 + col];   // A-row M = col for both lane halves

        // Layer 1: native packed f16 (v_pk_fma_f16 / v_pk_max_num_f16),
        // result lands directly in WMMA A layout.
        const _Float16 xh = (_Float16)x;
        const _Float16 zh = (_Float16)0.f;
        v16h a0, a1;
#pragma unroll
        for (int i = 0; i < 16; ++i) {
            _Float16 h0 = xh * w1h[0][i] + b1h[0][i];
            _Float16 h1 = xh * w1h[1][i] + b1h[1][i];
            a0[i] = h0 > zh ? h0 : zh;
            a1[i] = h1 > zh ? h1 : zh;
        }

        // Layer 2: 8 WMMAs, relu, spill f16 activations to LDS row-major [16][64]
#pragma unroll
        for (int nb = 0; nb < 4; ++nb) {
            v8f c = {};
            c = __builtin_amdgcn_wmma_f32_16x16x32_f16(false, a0, false, Bt2[0][nb],
                                                       (short)0, c, false, false);
            c = __builtin_amdgcn_wmma_f32_16x16x32_f16(false, a1, false, Bt2[1][nb],
                                                       (short)0, c, false, false);
#pragma unroll
            for (int v = 0; v < 8; ++v) {
                float h = fmaxf(c[v] + b2p[nb], 0.f);
                my[(v + 8 * hi) * 64 + 16 * nb + col] = (_Float16)h;
            }
        }

        // order this wave's DS store->load traffic (gfx1250 split counter)
        asm volatile("s_wait_dscnt 0" ::: "memory");

        // Reload in A layout: 4 contiguous 16B chunks -> ds_load_b128
        v16h a20, a21;
        {
            const _Float16* row = my + col * 64 + 8 * hi;
            v8h c0 = *(const v8h*)(row +  0);
            v8h c1 = *(const v8h*)(row + 16);
            v8h c2 = *(const v8h*)(row + 32);
            v8h c3 = *(const v8h*)(row + 48);
#pragma unroll
            for (int i = 0; i < 8; ++i) {
                a20[i] = c0[i]; a20[8 + i] = c1[i];
                a21[i] = c2[i]; a21[8 + i] = c3[i];
            }
        }

        // Layer 3: 4 WMMAs, relu
        float h3[2][8];
#pragma unroll
        for (int nb = 0; nb < 2; ++nb) {
            v8f c = {};
            c = __builtin_amdgcn_wmma_f32_16x16x32_f16(false, a20, false, Bt3[0][nb],
                                                       (short)0, c, false, false);
            c = __builtin_amdgcn_wmma_f32_16x16x32_f16(false, a21, false, Bt3[1][nb],
                                                       (short)0, c, false, false);
#pragma unroll
            for (int v = 0; v < 8; ++v) h3[nb][v] = fmaxf(c[v] + b3p[nb], 0.f);
        }

        // Layer 4: per-lane partials, butterfly reduce across 16-lane groups
        float part[8];
#pragma unroll
        for (int v = 0; v < 8; ++v) part[v] = h3[0][v] * w4a + h3[1][v] * w4b;
#pragma unroll
        for (int v = 0; v < 8; ++v) {
            part[v] += __shfl_xor(part[v], 1, 16);
            part[v] += __shfl_xor(part[v], 2, 16);
            part[v] += __shfl_xor(part[v], 4, 16);
            part[v] += __shfl_xor(part[v], 8, 16);
        }

        // writer lane (col<8) handles row m = 8*hi + (lane&7); fetch its x first
        const int m_all = 8 * hi + (lane & 7);
        const float xm  = __shfl(x, m_all, 16);    // all lanes active here
        if (col < 8) {
            float z = part[0];
#pragma unroll
            for (int v = 1; v < 8; ++v) if ((lane & 7) == v) z = part[v];
            float sg = 1.f / (1.f + __expf(-(z + b4s)));

            int e = t * 16 + m_all;
            int j = e & (NN - 1);
            int i = (e >> 9) & (NN - 1);
            int b = e >> 18;
            float mf = (nmask[b * NN + i] != 0 && nmask[b * NN + j] != 0) ? 1.f : 0.f;
            out[e] = rw * (sg * mf) + (1.f - rw) * xm;
        }
    }
}

// ---------------------------------------------------------------------------
// Kernel 2: in-place symmetrize + zero diagonal. Each (i<=j) pair owned by
// exactly one thread -> race-free read-both-then-write-both.
// ---------------------------------------------------------------------------
__global__ __launch_bounds__(256)
void sym_kernel(float* __restrict__ out)
{
    int e = blockIdx.x * blockDim.x + threadIdx.x;
    if (e >= NELEM) return;
    int j = e & (NN - 1);
    int i = (e >> 9) & (NN - 1);
    int b = e >> 18;
    if (j < i) return;
    if (j == i) { out[e] = 0.f; return; }
    int et = (b * NN + j) * NN + i;
    float a = out[e];
    float c = out[et];
    float mval = 0.5f * (a + c);
    out[e]  = mval;
    out[et] = mval;
}

extern "C" void kernel_launch(void* const* d_in, const int* in_sizes, int n_in,
                              void* d_out, int out_size, void* d_ws, size_t ws_size,
                              hipStream_t stream)
{
    (void)in_sizes; (void)n_in; (void)out_size; (void)d_ws; (void)ws_size;
    const float* sim  = (const float*)d_in[0];
    const int*   nmsk = (const int*)  d_in[1];
    const float* W1   = (const float*)d_in[2];
    const float* b1   = (const float*)d_in[3];
    const float* g1   = (const float*)d_in[4];
    const float* be1  = (const float*)d_in[5];
    const float* m1   = (const float*)d_in[6];
    const float* v1   = (const float*)d_in[7];
    const float* W2   = (const float*)d_in[8];
    const float* b2   = (const float*)d_in[9];
    const float* g2   = (const float*)d_in[10];
    const float* be2  = (const float*)d_in[11];
    const float* m2   = (const float*)d_in[12];
    const float* v2   = (const float*)d_in[13];
    const float* W3   = (const float*)d_in[14];
    const float* b3   = (const float*)d_in[15];
    const float* W4   = (const float*)d_in[16];
    const float* b4   = (const float*)d_in[17];
    const float* rw   = (const float*)d_in[18];
    float* out = (float*)d_out;

    mlp_wmma_kernel<<<WGS, 256, 0, stream>>>(sim, nmsk, W1, b1, g1, be1, m1, v1,
                                             W2, b2, g2, be2, m2, v2, W3, b3,
                                             W4, b4, rw, out);
    sym_kernel<<<NELEM / 256, 256, 0, stream>>>(out);
}
